// PolicyCoreRNN_51754355916999
// MI455X (gfx1250) — matrix operand order repaired and checked by reading, not compiled, not run on hardware
//
#include <hip/hip_runtime.h>

// ---------------- problem constants ----------------
#define Hh   512
#define Ii   512
#define Tt   64
#define Bb   1024
#define G4   2048   // 4*H

typedef __attribute__((ext_vector_type(16))) _Float16 v16h;
typedef __attribute__((ext_vector_type(8)))  _Float16 v8h;
typedef __attribute__((ext_vector_type(8)))  float    v8f;

__device__ __forceinline__ float sigf(float x) { return 1.0f / (1.0f + __expf(-x)); }

// ---------------- one-time weight prep: fp32 -> fp16 combined [4H][I+H], bias sum ----------------
__global__ void PolicyCoreRNN_wprep(const float* __restrict__ W_ih,
                                    const float* __restrict__ W_hh,
                                    const float* __restrict__ b_ih,
                                    const float* __restrict__ b_hh,
                                    _Float16* __restrict__ wh,
                                    float* __restrict__ bsum) {
  int idx = blockIdx.x * 256 + threadIdx.x;        // 0 .. 2048*1024-1
  int g = idx >> 10;
  int k = idx & 1023;
  float w = (k < Hh) ? W_ih[(size_t)g * Ii + k] : W_hh[(size_t)g * Hh + (k - Hh)];
  wh[idx] = (_Float16)w;
  if (idx < G4) bsum[idx] = b_ih[idx] + b_hh[idx];
}

// ---------------- per-step prep: masked h_{t-1} -> fp16 ----------------
__global__ void PolicyCoreRNN_hprep(const float* __restrict__ rnn_states,
                                    const float* __restrict__ dones,
                                    const float* __restrict__ h_stack,
                                    _Float16* __restrict__ hh,
                                    int t) {
  int idx = blockIdx.x * 256 + threadIdx.x;        // 0 .. B*H-1
  int b = idx >> 9;
  int j = idx & 511;
  float keep = (t == 0) ? 1.0f : (1.0f - dones[(size_t)b * Tt + t - 1]);
  float h = (t == 0) ? rnn_states[(size_t)b * (2 * Hh) + j]
                     : h_stack[((size_t)(t - 1) * Bb + b) * Hh + j];
  hh[idx] = (_Float16)(h * keep);
}

// ---------------- fused LSTM step: WMMA GEMM (M=1024,N=2048,K=1024) + cell epilogue ----------------
// grid = (8, 16): blockIdx.x -> 128-row batch tile, blockIdx.y -> 32 gate-columns (of 512)
// block = 256 threads = 8 waves arranged 4(M: 32 rows each) x 2(N: 16 cols each);
// each wave owns 32 rows x 16 cols x 4 gates = 8 accumulators.
__global__ __launch_bounds__(256)
void PolicyCoreRNN_step(const float* __restrict__ head,        // (B*T, I) fp32
                        const float* __restrict__ rnn_states,  // (B, 2H)
                        const float* __restrict__ dones,       // (B*T,)
                        const _Float16* __restrict__ wh,       // (4H, 1024) fp16
                        const float* __restrict__ bsum,        // (4H,)
                        const _Float16* __restrict__ hh,       // (B, H) masked fp16
                        float* __restrict__ x_out,             // (B*T, H)
                        float* __restrict__ states,            // (B, 2H)
                        float* __restrict__ h_stack,           // (T, B, H)
                        float* __restrict__ c_stack,           // (T, B, H)
                        int t) {
  // double-buffered tiles, rows padded to 40 halves (20-dword stride -> conflict-free b128)
  __shared__ _Float16 As[2][128][40];       // 128 batch rows x 32 K-halves
  __shared__ _Float16 Bs[2][4][32][40];     // per gate: 32 cols x 32 K-halves

  const int tid   = threadIdx.x;
  const int lane  = tid & 31;
  const int wave  = tid >> 5;
  const int lr    = lane & 15;
  const int hiL   = lane >> 4;
  const int waveM = wave & 3;               // 4 M-groups of 32 rows
  const int waveN = wave >> 2;              // 2 N-groups of 16 cols

  const int bM0   = blockIdx.x * 128;       // batch-row base
  const int jBase = blockIdx.y * 32;        // gate-column base (within 512)
  const int m0    = waveM * 32;             // wave's 32-row sub-tile
  const int nb    = waveN * 16;             // wave's 16-col sub-tile within B tile

  v8f acc[4][2] = {};                       // [gate][m-subtile]

  // ---- staging: A tile (x part: fp32->fp16 manual; h part: fp16 async-to-LDS)
  auto stageA = [&](int buf, int kc) {
    const int k0 = kc * 32;
    if (kc < 16) {
      #pragma unroll
      for (int it = 0; it < 2; ++it) {
        int seg = tid + it * 256;           // 512 segments: (row, 8-half piece)
        int r = seg >> 2, s = seg & 3;
        int b = bM0 + r;
        const float* src = head + ((size_t)b * Tt + t) * Ii + k0 + s * 8;
        float4 f0 = *(const float4*)(src);
        float4 f1 = *(const float4*)(src + 4);
        v8h h8;
        h8[0] = (_Float16)f0.x; h8[1] = (_Float16)f0.y;
        h8[2] = (_Float16)f0.z; h8[3] = (_Float16)f0.w;
        h8[4] = (_Float16)f1.x; h8[5] = (_Float16)f1.y;
        h8[6] = (_Float16)f1.z; h8[7] = (_Float16)f1.w;
        *(v8h*)&As[buf][r][s * 8] = h8;
      }
    } else {
      #pragma unroll
      for (int it = 0; it < 2; ++it) {
        int seg = tid + it * 256;
        int r = seg >> 2, s = seg & 3;
        int b = bM0 + r;
        const _Float16* gp = hh + (size_t)b * Hh + (k0 - Hh) + s * 8;
        unsigned lo = (unsigned)(uintptr_t)&As[buf][r][s * 8];  // LDS byte offset
        asm volatile("global_load_async_to_lds_b128 %0, %1, off"
                     :: "v"(lo), "v"(gp) : "memory");
      }
    }
  };

  // ---- staging: B tiles (fp16 weights, async-to-LDS)
  auto stageB = [&](int buf, int kc) {
    const int k0 = kc * 32;
    #pragma unroll
    for (int it = 0; it < 2; ++it) {
      int seg = tid + it * 256;             // 512 segments: gate x col x piece
      int q = seg >> 7, n = (seg >> 2) & 31, s = seg & 3;
      const _Float16* gp = wh + (size_t)(q * Hh + jBase + n) * 1024 + k0 + s * 8;
      unsigned lo = (unsigned)(uintptr_t)&Bs[buf][q][n][s * 8];
      asm volatile("global_load_async_to_lds_b128 %0, %1, off"
                   :: "v"(lo), "v"(gp) : "memory");
    }
  };

  stageA(0, 0);
  stageB(0, 0);

  for (int kc = 0; kc < 32; ++kc) {         // K = 1024 in chunks of 32
    const int buf = kc & 1;
    asm volatile("s_wait_asynccnt 0x0" ::: "memory");  // my async writes landed in LDS
    __syncthreads();                                   // ...published to all waves

    if (kc + 1 < 32) {                       // prefetch next chunk into other buffer
      stageA(buf ^ 1, kc + 1);
      stageB(buf ^ 1, kc + 1);
    }

    // ---- A fragments (ISA 16-bit A 16x32 layout): halves 0..7 = K kb..kb+7, 8..15 = K kb+16..+23
    const int kbA = hiL * 8;
    v16h a[2];
    #pragma unroll
    for (int sm = 0; sm < 2; ++sm) {
      v8h alo = *(const v8h*)&As[buf][m0 + sm * 16 + lr][kbA];
      v8h ahi = *(const v8h*)&As[buf][m0 + sm * 16 + lr][kbA + 16];
      #pragma unroll
      for (int i = 0; i < 8; ++i) { a[sm][i] = alo[i]; a[sm][i + 8] = ahi[i]; }
    }

    // ---- B fragments (same K order as A): halves 0..7 = K kb..kb+7, 8..15 = K kb+16..+23
    const int kbB = hiL * 8;
    #pragma unroll
    for (int q = 0; q < 4; ++q) {
      v8h blo = *(const v8h*)&Bs[buf][q][nb + lr][kbB];
      v8h bhi = *(const v8h*)&Bs[buf][q][nb + lr][kbB + 16];
      v16h bfr;
      #pragma unroll
      for (int i = 0; i < 8; ++i) { bfr[i] = blo[i]; bfr[i + 8] = bhi[i]; }
      acc[q][0] = __builtin_amdgcn_wmma_f32_16x16x32_f16(
          false, a[0], false, bfr, (short)0, acc[q][0], false, false);
      acc[q][1] = __builtin_amdgcn_wmma_f32_16x16x32_f16(
          false, a[1], false, bfr, (short)0, acc[q][1], false, false);
    }
  }

  // ---- LSTM cell epilogue: C-layout lane L: VGPR r -> M = r + 8*(L/16), N = L%16
  const int j = jBase + nb + lr;
  const float bs_i = bsum[j];
  const float bs_f = bsum[Hh + j];
  const float bs_g = bsum[2 * Hh + j];
  const float bs_o = bsum[3 * Hh + j];

  #pragma unroll
  for (int sm = 0; sm < 2; ++sm) {
    #pragma unroll
    for (int r = 0; r < 8; ++r) {
      int b = bM0 + m0 + sm * 16 + r + hiL * 8;
      float keep = (t == 0) ? 1.0f : (1.0f - dones[(size_t)b * Tt + t - 1]);
      float cp = ((t == 0) ? rnn_states[(size_t)b * (2 * Hh) + Hh + j]
                           : c_stack[((size_t)(t - 1) * Bb + b) * Hh + j]) * keep;
      float ig = sigf(acc[0][sm][r] + bs_i);
      float fg = sigf(acc[1][sm][r] + bs_f);
      float gg = tanhf(acc[2][sm][r] + bs_g);
      float og = sigf(acc[3][sm][r] + bs_o);
      float c = fg * cp + ig * gg;
      float hv = og * tanhf(c);
      size_t so = ((size_t)t * Bb + b) * Hh + j;
      h_stack[so] = hv;
      c_stack[so] = c;
      x_out[((size_t)b * Tt + t) * Hh + j] = hv;
      if (t == Tt - 1) {
        states[(size_t)b * (2 * Hh) + j] = hv;
        states[(size_t)b * (2 * Hh) + Hh + j] = c;
      }
    }
  }
}

extern "C" void kernel_launch(void* const* d_in, const int* in_sizes, int n_in,
                              void* d_out, int out_size, void* d_ws, size_t ws_size,
                              hipStream_t stream) {
  const float* head  = (const float*)d_in[0];
  const float* rnn   = (const float*)d_in[1];
  const float* dones = (const float*)d_in[2];
  const float* W_ih  = (const float*)d_in[3];
  const float* W_hh  = (const float*)d_in[4];
  const float* b_ih  = (const float*)d_in[5];
  const float* b_hh  = (const float*)d_in[6];

  // workspace: fp16 combined weights (4 MB) + fused bias (8 KB) + masked fp16 h (1 MB)
  char* ws = (char*)d_ws;
  _Float16* wh   = (_Float16*)ws;
  float*    bsum = (float*)(ws + (size_t)G4 * 1024 * 2);
  _Float16* hh   = (_Float16*)(ws + (size_t)G4 * 1024 * 2 + G4 * 4);

  // d_out layout: x (B*T*H) | new_rnn_states (B*2H) | h_stack (T*B*H) | c_stack (T*B*H)
  float* out     = (float*)d_out;
  float* x_out   = out;
  float* states  = out + (size_t)Bb * Tt * Hh;
  float* h_stack = states + (size_t)Bb * 2 * Hh;
  float* c_stack = h_stack + (size_t)Tt * Bb * Hh;

  PolicyCoreRNN_wprep<<<(G4 * 1024) / 256, 256, 0, stream>>>(W_ih, W_hh, b_ih, b_hh, wh, bsum);

  for (int t = 0; t < Tt; ++t) {
    PolicyCoreRNN_hprep<<<(Bb * Hh) / 256, 256, 0, stream>>>(rnn, dones, h_stack, hh, t);
    PolicyCoreRNN_step<<<dim3(8, 16), 256, 0, stream>>>(
        head, rnn, dones, wh, bsum, hh, x_out, states, h_stack, c_stack, t);
  }
}
